// RecurrentHyenaBlock_29549374996512
// MI455X (gfx1250) — matrix operand order repaired and checked by reading, not compiled, not run on hardware
//
#include <hip/hip_runtime.h>
#include <hip/hip_bf16.h>

typedef __bf16 bf16;
typedef __attribute__((ext_vector_type(16))) __bf16 v16bf;
typedef __attribute__((ext_vector_type(8)))  __bf16 v8bf;
typedef __attribute__((ext_vector_type(8)))  float  v8f;

#define SLEN 2048
#define SDIM 1024
#define SDFF 4096
#define SB   8
#define FFTN 4096
#define LOG2N 12
#define NT   256
#define PI_F 3.14159265358979323846f

// ---------------------------------------------------------------- LayerNorm
template <typename OutT>
__global__ void ln_kernel(const float* __restrict__ x, const float* __restrict__ g,
                          const float* __restrict__ bb, OutT* __restrict__ y) {
  const int row = blockIdx.x;
  const int tid = threadIdx.x;
  __shared__ float s1[NT], s2[NT];
  const float* xr = x + (size_t)row * SDIM;
  float v[4];
  float lsum = 0.f, lsq = 0.f;
#pragma unroll
  for (int i = 0; i < 4; ++i) {
    float t = xr[tid + i * NT];
    v[i] = t; lsum += t; lsq += t * t;
  }
  s1[tid] = lsum; s2[tid] = lsq;
  __syncthreads();
  for (int off = NT / 2; off > 0; off >>= 1) {
    if (tid < off) { s1[tid] += s1[tid + off]; s2[tid] += s2[tid + off]; }
    __syncthreads();
  }
  const float mu  = s1[0] * (1.f / SDIM);
  const float var = s2[0] * (1.f / SDIM) - mu * mu;
  const float rs  = rsqrtf(var + 1e-5f);
  OutT* yr = y + (size_t)row * SDIM;
#pragma unroll
  for (int i = 0; i < 4; ++i) {
    int c = tid + i * NT;
    float o = (v[i] - mu) * rs * g[c] + bb[c];
    yr[c] = (OutT)o;
  }
}

// ------------------------------------------- weight transpose f32 -> bf16 [N][K]
template <int K, int N>
__global__ void wt_kernel(const float* __restrict__ w, bf16* __restrict__ wT) {
  const int idx = blockIdx.x * NT + threadIdx.x;   // total = K*N = 4M
  const int n = idx >> __builtin_ctz(K);           // K is power of two
  const int k = idx & (K - 1);
  wT[idx] = (bf16)w[(size_t)k * N + n];
}

__global__ void cvt_kernel(const float* __restrict__ x, bf16* __restrict__ y) {
  int idx = blockIdx.x * NT + threadIdx.x;
  y[idx] = (bf16)x[idx];
}

// ---------------------------------------------------------------- FFT machinery
// one-time global twiddle table: tw[i] = exp(-2*pi*i*j/N), interleaved (re,im)
__global__ void twiddle_init_kernel(float* __restrict__ tw) {
  const int i = blockIdx.x * NT + threadIdx.x;     // 2048 total
  float s, c;
  sincosf(-PI_F * (float)i * (2.f / (float)FFTN), &s, &c);
  tw[2 * i]     = c;
  tw[2 * i + 1] = s;
}

__device__ __forceinline__ void fft_lds_twiddles(const float* __restrict__ twg,
                                                 float* twr, float* twi) {
  const float2* t2 = (const float2*)twg;
#pragma unroll
  for (int u = 0; u < (FFTN / 2) / NT; ++u) {
    int i = threadIdx.x + u * NT;
    float2 w = t2[i];
    twr[i] = w.x; twi[i] = w.y;
  }
}

// forward radix-2 DIF, natural-order input -> bit-reversed-order spectrum (in place)
__device__ __forceinline__ void fft_dif(float* xr, float* xi,
                                        const float* twr, const float* twi) {
#pragma unroll
  for (int s = 0; s < LOG2N; ++s) {
    const int l = (FFTN / 2) >> s;                 // compile-time per stage
    __syncthreads();
    for (int t = threadIdx.x; t < FFTN / 2; t += NT) {
      const int j = t & (l - 1);
      const int p = 2 * t - j;
      const int wix = j << s;                      // j * (N/(2l))
      float ur = xr[p],     ui = xi[p];
      float vr = xr[p + l], vi = xi[p + l];
      float wr = twr[wix],  wi = twi[wix];
      xr[p] = ur + vr; xi[p] = ui + vi;
      float dr = ur - vr, di = ui - vi;
      xr[p + l] = dr * wr - di * wi;
      xi[p + l] = dr * wi + di * wr;
    }
  }
  __syncthreads();
}

// inverse radix-2 DIT, bit-reversed-order input -> natural order, unscaled (x N)
__device__ __forceinline__ void fft_dit_inv(float* xr, float* xi,
                                            const float* twr, const float* twi) {
#pragma unroll
  for (int s = 0; s < LOG2N; ++s) {
    const int l = 1 << s;                          // compile-time per stage
    __syncthreads();
    for (int t = threadIdx.x; t < FFTN / 2; t += NT) {
      const int j = t & (l - 1);
      const int p = 2 * t - j;
      const int wix = j << (LOG2N - 1 - s);        // j * (N/(2l))
      float wr = twr[wix], wi = -twi[wix];         // conjugate
      float br = xr[p + l], bi = xi[p + l];
      float vr = br * wr - bi * wi;
      float vi = br * wi + bi * wr;
      float ur = xr[p], ui = xi[p];
      xr[p]     = ur + vr; xi[p]     = ui + vi;
      xr[p + l] = ur - vr; xi[p + l] = ui - vi;
    }
  }
  __syncthreads();
}

// spectrum of window[t]*h[t,d] per channel d (kept in bit-reversed order)
__global__ void fft_filter_kernel(const float* __restrict__ h, const float* __restrict__ a,
                                  const float* __restrict__ twg,
                                  float* __restrict__ wfh_re, float* __restrict__ wfh_im) {
  __shared__ float xr[FFTN], xi[FFTN], twr[FFTN / 2], twi[FFTN / 2];
  const int d = blockIdx.x;
  fft_lds_twiddles(twg, twr, twi);
  const float decay = __expf(a[0]);
  for (int t = threadIdx.x; t < FFTN; t += NT) {
    float v = 0.f;
    if (t < SLEN) v = __expf(-(float)t * decay) * h[(size_t)t * SDIM + d];
    xr[t] = v; xi[t] = 0.f;
  }
  fft_dif(xr, xi, twr, twi);
  float* oR = wfh_re + (size_t)d * FFTN;
  float* oI = wfh_im + (size_t)d * FFTN;
  for (int t = threadIdx.x; t < FFTN; t += NT) { oR[t] = xr[t]; oI[t] = xi[t]; }
}

// fused: FFT(x_ln channel) * filter-spectrum -> iFFT -> [:LEN] + hidden_init
__global__ void fft_conv_kernel(const float* __restrict__ xln,
                                const float* __restrict__ wfh_re,
                                const float* __restrict__ wfh_im,
                                const float* __restrict__ twg,
                                const float* __restrict__ hid0,
                                float* __restrict__ out) {
  __shared__ float xr[FFTN], xi[FFTN], twr[FFTN / 2], twi[FFTN / 2];
  const int d = blockIdx.x & (SDIM - 1);
  const int b = blockIdx.x >> 10;
  fft_lds_twiddles(twg, twr, twi);
  const float* src = xln + (size_t)b * SLEN * SDIM + d;
  for (int t = threadIdx.x; t < FFTN; t += NT) {
    xr[t] = (t < SLEN) ? src[(size_t)t * SDIM] : 0.f;
    xi[t] = 0.f;
  }
  fft_dif(xr, xi, twr, twi);
  const float* fR = wfh_re + (size_t)d * FFTN;
  const float* fI = wfh_im + (size_t)d * FFTN;
  for (int t = threadIdx.x; t < FFTN; t += NT) {
    float ar = xr[t], ai = xi[t];
    float br = fR[t], bi = fI[t];
    xr[t] = ar * br - ai * bi;
    xi[t] = ar * bi + ai * br;
  }
  fft_dit_inv(xr, xi, twr, twi);
  float* dst = out + (size_t)b * SLEN * SDIM + d;
  const float* hs = hid0 + d;
  for (int t = threadIdx.x; t < SLEN; t += NT) {
    dst[(size_t)t * SDIM] = xr[t] * (1.f / (float)FFTN) + hs[(size_t)t * SDIM];
  }
}

// ---------------------------------------------------------------- WMMA GEMM
struct bfpair { v8bf lo, hi; };

__device__ __forceinline__ v16bf load_a_frag(const bf16* __restrict__ p) {
  bfpair t;
  t.lo = *(const v8bf*)(p);        // K = klo .. klo+7
  t.hi = *(const v8bf*)(p + 16);   // K = klo+16 .. klo+23
  return __builtin_bit_cast(v16bf, t);
}
__device__ __forceinline__ v16bf load_b_frag(const bf16* __restrict__ p) {
  bfpair t;
  t.lo = *(const v8bf*)(p);        // K = klo .. klo+7
  t.hi = *(const v8bf*)(p + 16);   // K = klo+16 .. klo+23
  return __builtin_bit_cast(v16bf, t);
}

// C = act(A[M,K] @ BT[N,K]^T + bias) (+resid). Block: 256 thr = 8 waves,
// wave tile 32x64, block tile 64x256. All dims divide exactly.
template <int ACT, typename OutT>
__global__ void gemm_kernel(const bf16* __restrict__ A, const bf16* __restrict__ BT,
                            const float* __restrict__ bias, OutT* __restrict__ out,
                            const float* __restrict__ resid, int M, int N, int K) {
  const int lane = threadIdx.x & 31;
  const int wave = threadIdx.x >> 5;
  const int wm = wave >> 2;             // 0..1
  const int wn = wave & 3;              // 0..3
  const int m_base = blockIdx.y * 64 + wm * 32;
  const int n_base = blockIdx.x * 256 + wn * 64;
  const int lr = lane & 15;
  const int lh = lane >> 4;

  v8f acc[2][4] = {};

  const bf16* a0 = A + (size_t)(m_base + lr) * K + (lh << 3);
  const bf16* a1 = a0 + (size_t)16 * K;
  const bf16* bp[4];
#pragma unroll
  for (int nt = 0; nt < 4; ++nt)
    bp[nt] = BT + (size_t)(n_base + nt * 16 + lr) * K + (lh << 3);

  for (int k = 0; k < K; k += 32) {
    v16bf af[2], bf[4];
    af[0] = load_a_frag(a0 + k);
    af[1] = load_a_frag(a1 + k);
#pragma unroll
    for (int nt = 0; nt < 4; ++nt) bf[nt] = load_b_frag(bp[nt] + k);
#pragma unroll
    for (int mt = 0; mt < 2; ++mt)
#pragma unroll
      for (int nt = 0; nt < 4; ++nt)
        acc[mt][nt] = __builtin_amdgcn_wmma_f32_16x16x32_bf16(
            false, af[mt], false, bf[nt], (short)0, acc[mt][nt], false, false);
  }

  const int colb = n_base + lr;
  const int rowb = m_base + (lh << 3);
#pragma unroll
  for (int mt = 0; mt < 2; ++mt) {
#pragma unroll
    for (int nt = 0; nt < 4; ++nt) {
      const int col = colb + nt * 16;
      const float bv = bias[col];
#pragma unroll
      for (int r = 0; r < 8; ++r) {
        const int row = rowb + mt * 16 + r;
        float v = acc[mt][nt][r] + bv;
        if (ACT == 1) v = v * (1.f / (1.f + __expf(-v)));   // SiLU
        const size_t o = (size_t)row * N + col;
        if (resid) v += resid[o];
        out[o] = (OutT)v;
      }
    }
  }
}

// ---------------------------------------------------------------- launch
extern "C" void kernel_launch(void* const* d_in, const int* in_sizes, int n_in,
                              void* d_out, int out_size, void* d_ws, size_t ws_size,
                              hipStream_t stream) {
  (void)in_sizes; (void)n_in; (void)out_size; (void)ws_size;
  const float* z    = (const float*)d_in[0];
  const float* pos  = (const float*)d_in[1];
  const float* a    = (const float*)d_in[2];
  const float* hid0 = (const float*)d_in[3];
  const float* ln_g = (const float*)d_in[4];
  const float* ln_b = (const float*)d_in[5];
  const float* wp1  = (const float*)d_in[6];
  const float* bp1  = (const float*)d_in[7];
  const float* wp2  = (const float*)d_in[8];
  const float* bp2  = (const float*)d_in[9];
  const float* w1   = (const float*)d_in[10];
  const float* b1   = (const float*)d_in[11];
  const float* w2   = (const float*)d_in[12];
  const float* b2   = (const float*)d_in[13];
  float* out = (float*)d_out;

  char* ws = (char*)d_ws;
  size_t off = 0;
  auto wsalloc = [&](size_t bytes) -> void* {
    void* p = ws + off;
    off += (bytes + 255) & ~(size_t)255;
    return p;
  };
  float* x_ln   = (float*)wsalloc((size_t)SB * SLEN * SDIM * 4);   // 64 MB
  float* wfh_re = (float*)wsalloc((size_t)SDIM * FFTN * 4);        // 16 MB
  float* wfh_im = (float*)wsalloc((size_t)SDIM * FFTN * 4);        // 16 MB
  float* twg    = (float*)wsalloc((size_t)FFTN * 4);               // 16 KB (re,im interleaved)
  bf16*  wp1T   = (bf16*) wsalloc((size_t)SDIM * SDFF * 2);        // 8 MB
  bf16*  wp2T   = (bf16*) wsalloc((size_t)SDFF * SDIM * 2);        // 8 MB
  bf16*  w1T    = (bf16*) wsalloc((size_t)SDIM * SDFF * 2);        // 8 MB
  bf16*  w2T    = (bf16*) wsalloc((size_t)SDFF * SDIM * 2);        // 8 MB
  bf16*  posb   = (bf16*) wsalloc((size_t)SLEN * SDIM * 2);        // 4 MB
  bf16*  hidp   = (bf16*) wsalloc((size_t)SLEN * SDFF * 2);        // 16 MB
  float* hcoef  = (float*)wsalloc((size_t)SLEN * SDIM * 4);        // 8 MB
  bf16*  zlnb   = (bf16*) wsalloc((size_t)SB * SLEN * SDIM * 2);   // 32 MB
  bf16*  hidb   = (bf16*) wsalloc((size_t)SB * SLEN * SDFF * 2);   // 128 MB

  const int wtb = (SDIM * SDFF) / NT;
  wt_kernel<SDIM, SDFF><<<wtb, NT, 0, stream>>>(wp1, wp1T);
  wt_kernel<SDFF, SDIM><<<wtb, NT, 0, stream>>>(wp2, wp2T);
  wt_kernel<SDIM, SDFF><<<wtb, NT, 0, stream>>>(w1,  w1T);
  wt_kernel<SDFF, SDIM><<<wtb, NT, 0, stream>>>(w2,  w2T);
  cvt_kernel<<<(SLEN * SDIM) / NT, NT, 0, stream>>>(pos, posb);
  twiddle_init_kernel<<<(FFTN / 2) / NT, NT, 0, stream>>>(twg);

  // x_ln = layer_norm(z)
  ln_kernel<float><<<SB * SLEN, NT, 0, stream>>>(z, ln_g, ln_b, x_ln);

  // h = ffn(pos)
  gemm_kernel<1, bf16><<<dim3(SDFF / 256, SLEN / 64), NT, 0, stream>>>(
      posb, wp1T, bp1, hidp, nullptr, SLEN, SDFF, SDIM);
  gemm_kernel<0, float><<<dim3(SDIM / 256, SLEN / 64), NT, 0, stream>>>(
      hidp, wp2T, bp2, hcoef, nullptr, SLEN, SDIM, SDFF);

  // spectral filter, then fused FFT-conv + hidden residual -> znew in d_out
  fft_filter_kernel<<<SDIM, NT, 0, stream>>>(hcoef, a, twg, wfh_re, wfh_im);
  fft_conv_kernel<<<SB * SDIM, NT, 0, stream>>>(x_ln, wfh_re, wfh_im, twg, hid0, out);

  // out = ffn(layer_norm(znew)) + znew
  ln_kernel<bf16><<<SB * SLEN, NT, 0, stream>>>(out, ln_g, ln_b, zlnb);
  gemm_kernel<1, bf16><<<dim3(SDFF / 256, (SB * SLEN) / 64), NT, 0, stream>>>(
      zlnb, w1T, b1, hidb, nullptr, SB * SLEN, SDFF, SDIM);
  gemm_kernel<0, float><<<dim3(SDIM / 256, (SB * SLEN) / 64), NT, 0, stream>>>(
      hidb, w2T, b2, out, out, SB * SLEN, SDIM, SDFF);
}
